// GAT_layer_14809047236736
// MI455X (gfx1250) — matrix operand order repaired and checked by reading, not compiled, not run on hardware
//
#include <hip/hip_runtime.h>

typedef __attribute__((ext_vector_type(2)))  float v2f;
typedef __attribute__((ext_vector_type(4)))  float f32x4;
typedef __attribute__((ext_vector_type(8)))  float v8f;
typedef __attribute__((ext_vector_type(4)))  int   i32x4;
typedef __attribute__((ext_vector_type(8)))  __bf16 v8bf;
typedef __attribute__((ext_vector_type(16))) __bf16 v16bf;
typedef __attribute__((ext_vector_type(8)))  unsigned short u16x8;

#define ALPHA 0.2f
#define NEGINF (-1.0e15f)
#define LOG2E 1.44269504088896f

__device__ __forceinline__ float fast_exp2(float x) {
#if __has_builtin(__builtin_amdgcn_exp2f)
  return __builtin_amdgcn_exp2f(x);
#else
  return exp2f(x);
#endif
}

__device__ __forceinline__ unsigned short f32_to_bf16(float f) {
  unsigned u = __builtin_bit_cast(unsigned, f);
  return (unsigned short)((u + 0x7FFFu + ((u >> 16) & 1u)) >> 16);  // RNE
}

// ---------------- Kernel 1: h_hat = h @ W^T + b (WMMA f32 16x16x4) ----------
// block = 128 threads = 4 waves; block computes rows [i0,i0+16) x all 64 cols.
// wave w owns output cols [16w,16w+16). Also emits bf16 transposed copy
// hhatT[64][8192] so the flash kernel's B operand is two contiguous 16B loads/lane.
__global__ __launch_bounds__(128) void k_hhat(
    const float* __restrict__ h, const float* __restrict__ Ww,
    const float* __restrict__ Wb, float* __restrict__ hhat,
    unsigned short* __restrict__ hhatT) {
  const int tid  = threadIdx.x;
  const int wave = tid >> 5, lane = tid & 31;
  const int i0   = blockIdx.x * 16;
  const int l16  = lane & 15, hh = lane >> 4;
  const int ncol = wave * 16 + l16;          // B operand: n = lane%16

  v8f acc = {};
  for (int k0 = 0; k0 < 256; k0 += 4) {
    const int ka = k0 + hh * 2;              // A/B: k = 2*(lane/16) + {0,1}
    v2f a = *(const v2f*)(h  + (size_t)(i0 + l16) * 256 + ka);
    v2f b = *(const v2f*)(Ww + (size_t)ncol       * 256 + ka);
    acc = __builtin_amdgcn_wmma_f32_16x16x4_f32(false, a, false, b,
                                                (short)0, acc, false, false);
  }
  const float bias = Wb[ncol];
#pragma unroll
  for (int j = 0; j < 8; ++j) {              // C: row = j + 8*(lane/16), col = lane%16
    float v  = acc[j] + bias;
    int row  = i0 + j + 8 * hh;
    hhat[(size_t)row * 64 + ncol] = v;
    hhatT[(size_t)ncol * 8192 + row] = f32_to_bf16(v);
  }
}

// ---------------- Kernel 2: s1/s2 = h_hat @ a1/a2 ---------------------------
__global__ __launch_bounds__(256) void k_svec(
    const float* __restrict__ hhat, const float* __restrict__ aw,
    float* __restrict__ s1, float* __restrict__ s2) {
  int i = blockIdx.x * blockDim.x + threadIdx.x;   // 8192 rows
  float a1 = 0.f, a2 = 0.f;
  const float* r = hhat + (size_t)i * 64;
#pragma unroll 16
  for (int n = 0; n < 64; ++n) { float v = r[n]; a1 += v * aw[n]; a2 += v * aw[64 + n]; }
  s1[i] = a1; s2[i] = a2;
}

// ---------------- Kernel 3: fused flash GAT softmax + P@h_hat ---------------
// block = 128 thr (4 waves), owns 16 query rows; streams all 8192 cols in
// tiles of 64. The 16x64 int32 mask tile is the dominant HBM stream, so it is
// register double-buffered one tile ahead (branchless, index wraps) and
// global_prefetch'ed 4 tiles ahead into L2.
__global__ __launch_bounds__(128) void k_flash(
    const int* __restrict__ mask, const float* __restrict__ s1v,
    const float* __restrict__ s2v, const float* __restrict__ abp,
    const unsigned short* __restrict__ hhatT, float* __restrict__ out) {
  __shared__ __align__(16) unsigned short pbuf[16 * 64];  // bf16 P tile [i][j]
  __shared__ float scale_lds[16];
  __shared__ float linv[16];

  const int tid  = threadIdx.x;
  const int wave = tid >> 5, lane = tid & 31;
  const int i0   = blockIdx.x * 16;
  // score-phase role: row xi, cols [c0, c0+8)
  const int xi  = tid >> 3;
  const int grp = tid & 7, c0 = grp * 8;
  const float s1i = s1v[i0 + xi];
  const float ab0 = abp[0];
  float mrun = -3.0e38f, lsum = 0.f;
  // wmma-phase role
  const int l16 = lane & 31 & 15, hh = lane >> 4;
  v8f acc = {};
  const unsigned short* Bbase = hhatT + (size_t)(wave * 16 + l16) * 8192;
  const int* mrow = mask + (size_t)(i0 + xi) * 8192 + c0;

  // prime double buffer with tile 0
  i32x4 mk0 = ((const i32x4*)mrow)[0];
  i32x4 mk1 = ((const i32x4*)mrow)[1];
  f32x4 s2a = *(const f32x4*)(s2v + c0);
  f32x4 s2b = *(const f32x4*)(s2v + c0 + 4);

  for (int j0 = 0; j0 < 8192; j0 += 64) {
    // ---- prefetch: registers one tile ahead, L2 four tiles ahead ----
    const int jn = (j0 + 64)  & 8191;          // wraps harmlessly on last iter
    const int jf = (j0 + 256) & 8191;
    __builtin_prefetch(mrow + jf, 0, 1);       // -> global_prefetch_b8 (L2)
    i32x4 nk0 = ((const i32x4*)(mrow + jn))[0];
    i32x4 nk1 = ((const i32x4*)(mrow + jn))[1];
    f32x4 ns2a = *(const f32x4*)(s2v + jn + c0);
    f32x4 ns2b = *(const f32x4*)(s2v + jn + c0 + 4);

    __syncthreads();                           // pbuf free to overwrite
    float x[8];
#pragma unroll
    for (int q = 0; q < 8; ++q) {
      float s2j = (q < 4) ? s2a[q] : s2b[q - 4];
      float z = s1i + s2j + ab0;
      z = z > 0.f ? z : ALPHA * z;             // leaky_relu
      int mv = (q < 4) ? mk0[q] : mk1[q - 4];
      x[q] = (mv > 0) ? z : NEGINF;
    }
    float tmax = x[0];
#pragma unroll
    for (int q = 1; q < 8; ++q) tmax = fmaxf(tmax, x[q]);
    tmax = fmaxf(tmax, __shfl_xor(tmax, 1, 32));
    tmax = fmaxf(tmax, __shfl_xor(tmax, 2, 32));
    tmax = fmaxf(tmax, __shfl_xor(tmax, 4, 32));
    const float mnew = fmaxf(mrun, tmax);
    const float scl  = fast_exp2((mrun - mnew) * LOG2E);
    float psum = 0.f;
    u16x8 pv;
#pragma unroll
    for (int q = 0; q < 8; ++q) {
      float p = fast_exp2((x[q] - mnew) * LOG2E);
      psum += p;
      pv[q] = f32_to_bf16(p);
    }
    psum += __shfl_xor(psum, 1, 32);
    psum += __shfl_xor(psum, 2, 32);
    psum += __shfl_xor(psum, 4, 32);
    lsum = lsum * scl + psum;
    mrun = mnew;
    *(u16x8*)&pbuf[xi * 64 + c0] = pv;         // one b128 LDS store
    if (grp == 0) scale_lds[xi] = scl;
    // rotate double buffer (dead after this point in the iteration)
    mk0 = nk0; mk1 = nk1; s2a = ns2a; s2b = ns2b;
    __syncthreads();

    // rescale persistent accumulator: element j -> row j + 8*(lane/16)
#pragma unroll
    for (int j = 0; j < 8; ++j) acc[j] *= scale_lds[j + 8 * hh];

    // two K=32 slices of the 16x64 tile
#pragma unroll
    for (int s = 0; s < 2; ++s) {
      // A operand (16-bit A table): lane m=lane%16; v0..3 <- k = hh*8+{0..7},
      // v4..7 <- +16, all within slice base s*32.
      const unsigned short* ap = &pbuf[l16 * 64 + s * 32 + hh * 8];
      v8bf alo = *(const v8bf*)ap;
      v8bf ahi = *(const v8bf*)(ap + 16);
      v16bf A;
#pragma unroll
      for (int q = 0; q < 8; ++q) { A[q] = alo[q]; A[8 + q] = ahi[q]; }
      // B operand: n = lane%16, k = (lane/16)*8 + {0..7}, then +16 (same as A)
      // in the transposed bf16 copy (L2-resident, 1 MiB).
      const unsigned short* bp = Bbase + j0 + s * 32 + hh * 8;
      v8bf blo = *(const v8bf*)bp;
      v8bf bhi = *(const v8bf*)(bp + 16);
      v16bf B;
#pragma unroll
      for (int q = 0; q < 8; ++q) { B[q] = blo[q]; B[8 + q] = bhi[q]; }
      acc = __builtin_amdgcn_wmma_f32_16x16x32_bf16(false, A, false, B,
                                                    (short)0, acc, false, false);
    }
  }

  if (grp == 0) linv[xi] = 1.0f / lsum;        // reciprocal once per row
  __syncthreads();
#pragma unroll
  for (int j = 0; j < 8; ++j) {
    int row  = j + 8 * hh;
    float r  = acc[j] * linv[row];
    float e  = r > 0.f ? r : (fast_exp2(r * LOG2E) - 1.f);   // elu
    out[(size_t)(i0 + row) * 64 + wave * 16 + l16] = e;
  }
}

extern "C" void kernel_launch(void* const* d_in, const int* in_sizes, int n_in,
                              void* d_out, int out_size, void* d_ws, size_t ws_size,
                              hipStream_t stream) {
  const float* h    = (const float*)d_in[0];
  const int*   mask = (const int*)  d_in[1];
  const float* Ww   = (const float*)d_in[2];
  const float* Wb   = (const float*)d_in[3];
  const float* aw   = (const float*)d_in[4];
  const float* ab   = (const float*)d_in[5];
  float* out = (float*)d_out;

  char* ws = (char*)d_ws;
  float*          hhat  = (float*)(ws);                         // 2 MiB
  unsigned short* hhatT = (unsigned short*)(ws + 2097152);      // 1 MiB
  float*          s1    = (float*)(ws + 2097152 + 1048576);     // 32 KiB
  float*          s2    = (float*)(ws + 2097152 + 1048576 + 32768);

  k_hhat <<<512, 128, 0, stream>>>(h, Ww, Wb, hhat, hhatT);
  k_svec <<<32, 256, 0, stream>>>(hhat, aw, s1, s2);
  k_flash<<<512, 128, 0, stream>>>(mask, s1, s2, ab, hhatT, out);
}
